// LocalAttention_19000935317543
// MI455X (gfx1250) — matrix operand order repaired and checked by reading, not compiled, not run on hardware
//
#include <hip/hip_runtime.h>

// ---------------------------------------------------------------------------
// LocalAttention for MI455X (gfx1250, wave32, WMMA f16 16x16x32).
// Pipeline: f32->f16 weights | RMSNorm | QKV GEMM (WMMA) | split+RoPE(+V^T)
//           | flash attention (WMMA + online softmax) | out-proj GEMM (WMMA).
// ---------------------------------------------------------------------------

typedef __attribute__((ext_vector_type(16))) _Float16 v16h;
typedef __attribute__((ext_vector_type(8)))  _Float16 v8h;
typedef __attribute__((ext_vector_type(8)))  float    v8f;

#define D_MODEL 768
#define N_HEAD  12
#define HEAD_DIM 64
#define BATCH   4
#define HW      1024
#define NROWS   (BATCH * HW)          /* 4096 */
#define QKV_N   (3 * D_MODEL)         /* 2304 */
#define PPAD    72                    /* LDS row stride (halves) for P tile */

// A-fragment (16x32 f16, rows row-major, leading dim ld in halves).
// lane<16: K 0..7 & 16..23 ; lane>=16: K 8..15 & 24..31  (ISA 7.12.2)
static __device__ __forceinline__ v16h load_frag_a(const _Float16* base, int ld, int lane) {
    int r = lane & 15, hf = lane >> 4;
    const _Float16* p = base + r * ld + hf * 8;
    v8h lo = *(const v8h*)(p);
    v8h hi = *(const v8h*)(p + 16);
    v16h o;
#pragma unroll
    for (int i = 0; i < 8; ++i) { o[i] = lo[i]; o[i + 8] = hi[i]; }
    return o;
}

// B-fragment (32x16 f16) from W^T-form memory: row n holds K contiguous.
// lane: n = lane%16, lane<16: K 0..7 & 16..23 ; lane>=16: K 8..15 & 24..31.
static __device__ __forceinline__ v16h load_frag_b(const _Float16* base, int ld, int lane) {
    int n = lane & 15, hf = lane >> 4;
    const _Float16* p = base + n * ld + hf * 8;
    v8h lo = *(const v8h*)(p);
    v8h hi = *(const v8h*)(p + 16);
    v16h o;
#pragma unroll
    for (int i = 0; i < 8; ++i) { o[i] = lo[i]; o[i + 8] = hi[i]; }
    return o;
}

static __device__ __forceinline__ v8f wmma16(v16h a, v16h b, v8f c) {
    return __builtin_amdgcn_wmma_f32_16x16x32_f16(false, a, false, b, (short)0, c, false, false);
}

// ---------------------------------------------------------------------------
__global__ void f32_to_f16_kernel(const float* __restrict__ src,
                                  _Float16* __restrict__ dst, int n) {
    int i = blockIdx.x * 256 + threadIdx.x;
    if (i < n) dst[i] = (_Float16)src[i];
}

// ---------------------------------------------------------------------------
__global__ __launch_bounds__(256) void rmsnorm_kernel(const float* __restrict__ x,
                                                      const float* __restrict__ gamma,
                                                      _Float16* __restrict__ xn) {
    int row = blockIdx.x;
    const float* xr = x + (long)row * D_MODEL;
    float ss = 0.f;
    for (int c = threadIdx.x; c < D_MODEL; c += 256) { float v = xr[c]; ss += v * v; }
    __shared__ float red[256];
    red[threadIdx.x] = ss;
    __syncthreads();
    for (int s = 128; s > 0; s >>= 1) {
        if ((int)threadIdx.x < s) red[threadIdx.x] += red[threadIdx.x + s];
        __syncthreads();
    }
    float scale = rsqrtf(red[0] * (1.0f / D_MODEL) + 1e-6f);
    for (int c = threadIdx.x; c < D_MODEL; c += 256)
        xn[(long)row * D_MODEL + c] = (_Float16)(xr[c] * scale * gamma[c]);
}

// ---------------------------------------------------------------------------
// C(M,N) = A(M,K) @ W(N,K)^T ; f16 in, f32 accum; store f32 or f16.
// block = 8 waves; block tile 128x128; wave tile 32x64 (2x4 WMMA frags).
__global__ __launch_bounds__(256) void gemm_wt_kernel(const _Float16* __restrict__ A,
                                                      const _Float16* __restrict__ W,
                                                      float* __restrict__ Cf,
                                                      _Float16* __restrict__ Ch,
                                                      int M, int N, int K, int tilesN) {
    int lane = threadIdx.x & 31;
    int wave = threadIdx.x >> 5;
    int bx = blockIdx.x % tilesN;
    int by = blockIdx.x / tilesN;
    int m0 = by * 128 + (wave >> 1) * 32;
    int n0 = bx * 128 + (wave & 1) * 64;

    v8f acc[2][4] = {};
    for (int k0 = 0; k0 < K; k0 += 32) {
        v16h a0 = load_frag_a(A + (long)m0 * K + k0, K, lane);
        v16h a1 = load_frag_a(A + (long)(m0 + 16) * K + k0, K, lane);
#pragma unroll
        for (int nt = 0; nt < 4; ++nt) {
            v16h b = load_frag_b(W + (long)(n0 + nt * 16) * K + k0, K, lane);
            acc[0][nt] = wmma16(a0, b, acc[0][nt]);
            acc[1][nt] = wmma16(a1, b, acc[1][nt]);
        }
    }
    int cn = lane & 15, hf = lane >> 4;
#pragma unroll
    for (int mt = 0; mt < 2; ++mt)
#pragma unroll
        for (int nt = 0; nt < 4; ++nt)
#pragma unroll
            for (int r = 0; r < 8; ++r) {
                long row = m0 + mt * 16 + r + 8 * hf;
                long col = n0 + nt * 16 + cn;
                float v = acc[mt][nt][r];
                if (Cf) Cf[row * N + col] = v;
                else    Ch[row * N + col] = (_Float16)v;
            }
}

// ---------------------------------------------------------------------------
// qkv (f16, 4096 x 2304) -> Q,K (b,h,hw,hd) with axial RoPE ; V^T (b,h,hd,hw)
__global__ __launch_bounds__(256) void split_rope_kernel(const _Float16* __restrict__ qkv,
                                                         _Float16* __restrict__ Q,
                                                         _Float16* __restrict__ Kc,
                                                         _Float16* __restrict__ Vt) {
    int t = blockIdx.x * 256 + threadIdx.x;   // t < 4096*12*64
    int d = t & 63;
    int tmp = t >> 6;
    int h = tmp % 12;
    int m = tmp / 12;                          // b*1024 + i
    int i = m & 1023, b = m >> 10;
    int ri = i >> 5, ci = i & 31;
    int dh = d & 31;
    float pos = (dh < 16) ? (float)ri : (float)ci;
    // inv_freq = 10000^(-(dh&15)/16) ; ln(10000)/16 = 0.57564627...
    float th = pos * __expf(-(float)(dh & 15) * 0.5756462732485115f);
    float sn = __sinf(th), cs = __cosf(th);
    int dp = (d < 32) ? d + 32 : d - 32;
    float rs = (d < 32) ? -1.f : 1.f;

    long base = (long)m * QKV_N + h * 64;
    float q0 = (float)qkv[base + d],        q1 = (float)qkv[base + dp];
    float k0 = (float)qkv[base + 768 + d],  k1 = (float)qkv[base + 768 + dp];
    float vv = (float)qkv[base + 1536 + d];

    int bh = b * 12 + h;
    Q [((long)bh * HW + i) * 64 + d] = (_Float16)(q0 * cs + rs * q1 * sn);
    Kc[((long)bh * HW + i) * 64 + d] = (_Float16)(k0 * cs + rs * k1 * sn);
    Vt[((long)bh * 64 + d) * HW + i] = (_Float16)vv;
}

// ---------------------------------------------------------------------------
// Flash attention: one wave per 16 query rows of one (b,h).
// 8 waves/block -> 384 blocks. 3x3 neighborhood (incl. self) masked out.
__global__ __launch_bounds__(256) void attn_kernel(const _Float16* __restrict__ Q,
                                                   const _Float16* __restrict__ Kc,
                                                   const _Float16* __restrict__ Vt,
                                                   _Float16* __restrict__ O) {
    __shared__ _Float16 plds[8 * 16 * PPAD];
    int lane = threadIdx.x & 31;
    int wave = threadIdx.x >> 5;
    int gw = blockIdx.x * 8 + wave;
    int bh = gw >> 6;                 // 64 waves per (b,h)
    int i0 = (gw & 63) << 4;
    int b = bh / 12, h = bh % 12;
    const _Float16* Qb = Q  + (long)bh * HW * 64;
    const _Float16* Kb = Kc + (long)bh * HW * 64;
    const _Float16* Vb = Vt + (long)bh * 64 * HW;
    _Float16* pw = plds + wave * 16 * PPAD;

    v16h qa0 = load_frag_a(Qb + i0 * 64, 64, lane);
    v16h qa1 = load_frag_a(Qb + i0 * 64 + 32, 64, lane);

    v8f acc[4] = {};
    float mrow[8], lrow[8];
#pragma unroll
    for (int r = 0; r < 8; ++r) { mrow[r] = -1e30f; lrow[r] = 0.f; }

    int hf = lane >> 4, cn = lane & 15;
    int ri[8], ci[8];
#pragma unroll
    for (int r = 0; r < 8; ++r) {
        int i = i0 + r + 8 * hf;
        ri[r] = i >> 5; ci[r] = i & 31;
    }
    const float scale = 0.125f;       // 1/sqrt(64)

    for (int j0 = 0; j0 < HW; j0 += 64) {
        v8f sc[4];
#pragma unroll
        for (int jt = 0; jt < 4; ++jt) {
            v16h kb0 = load_frag_b(Kb + (long)(j0 + jt * 16) * 64, 64, lane);
            v16h kb1 = load_frag_b(Kb + (long)(j0 + jt * 16) * 64 + 32, 64, lane);
            v8f s = {};
            s = wmma16(qa0, kb0, s);
            s = wmma16(qa1, kb1, s);
            int j = j0 + jt * 16 + cn;
            int rj = j >> 5, cj = j & 31;
#pragma unroll
            for (int r = 0; r < 8; ++r) {
                int dr = ri[r] - rj; if (dr < 0) dr = -dr;
                int dc = ci[r] - cj; if (dc < 0) dc = -dc;
                sc[jt][r] = (dr <= 1 && dc <= 1) ? -1e30f : s[r] * scale;
            }
        }
        // online softmax update (row lives on a 16-lane half)
#pragma unroll
        for (int r = 0; r < 8; ++r) {
            float tmax = fmaxf(fmaxf(sc[0][r], sc[1][r]), fmaxf(sc[2][r], sc[3][r]));
#pragma unroll
            for (int off = 1; off < 16; off <<= 1)
                tmax = fmaxf(tmax, __shfl_xor(tmax, off, 32));
            float mnew = fmaxf(mrow[r], tmax);
            float alpha = __expf(mrow[r] - mnew);
            mrow[r] = mnew;
            lrow[r] *= alpha;
#pragma unroll
            for (int n = 0; n < 4; ++n) acc[n][r] *= alpha;
            float rsum = 0.f;
#pragma unroll
            for (int jt = 0; jt < 4; ++jt) {
                float p = __expf(sc[jt][r] - mnew);
                sc[jt][r] = p;
                rsum += p;
            }
#pragma unroll
            for (int off = 1; off < 16; off <<= 1)
                rsum += __shfl_xor(rsum, off, 32);
            lrow[r] += rsum;
        }
        // C-layout P -> row-major LDS tile (16 x 64, padded) -> A-fragments
#pragma unroll
        for (int jt = 0; jt < 4; ++jt)
#pragma unroll
            for (int r = 0; r < 8; ++r)
                pw[(r + 8 * hf) * PPAD + jt * 16 + cn] = (_Float16)sc[jt][r];
        asm volatile("s_wait_dscnt 0" ::: "memory");
        v16h pa0 = load_frag_a(pw, PPAD, lane);
        v16h pa1 = load_frag_a(pw + 32, PPAD, lane);
#pragma unroll
        for (int n = 0; n < 4; ++n) {
            v16h vb0 = load_frag_b(Vb + (long)(n * 16) * HW + j0, HW, lane);
            v16h vb1 = load_frag_b(Vb + (long)(n * 16) * HW + j0 + 32, HW, lane);
            acc[n] = wmma16(pa0, vb0, acc[n]);
            acc[n] = wmma16(pa1, vb1, acc[n]);
        }
        if (j0 + 64 < HW) __builtin_prefetch(Kb + (long)(j0 + 64) * 64, 0, 0);
    }
    // normalize + store heads-concatenated (b*HW+i, h*64+d) as f16
#pragma unroll
    for (int r = 0; r < 8; ++r) {
        float inv = 1.f / lrow[r];
        long row = (long)b * HW + i0 + r + 8 * hf;
#pragma unroll
        for (int n = 0; n < 4; ++n)
            O[row * D_MODEL + h * 64 + n * 16 + cn] = (_Float16)(acc[n][r] * inv);
    }
}

// ---------------------------------------------------------------------------
extern "C" void kernel_launch(void* const* d_in, const int* in_sizes, int n_in,
                              void* d_out, int out_size, void* d_ws, size_t ws_size,
                              hipStream_t stream) {
    const float* x     = (const float*)d_in[0];
    const float* w_qkv = (const float*)d_in[1];
    const float* w_out = (const float*)d_in[2];
    const float* gamma = (const float*)d_in[3];

    char* ws = (char*)d_ws;
    _Float16* xn   = (_Float16*)ws; ws += (size_t)NROWS * D_MODEL * 2;
    _Float16* wqh  = (_Float16*)ws; ws += (size_t)QKV_N * D_MODEL * 2;
    _Float16* woh  = (_Float16*)ws; ws += (size_t)D_MODEL * D_MODEL * 2;
    _Float16* qkvh = (_Float16*)ws; ws += (size_t)NROWS * QKV_N * 2;
    _Float16* Qb   = (_Float16*)ws; ws += (size_t)NROWS * D_MODEL * 2;
    _Float16* Kb   = (_Float16*)ws; ws += (size_t)NROWS * D_MODEL * 2;
    _Float16* Vtb  = (_Float16*)ws; ws += (size_t)NROWS * D_MODEL * 2;
    _Float16* Ob   = (_Float16*)ws; ws += (size_t)NROWS * D_MODEL * 2;

    f32_to_f16_kernel<<<(QKV_N * D_MODEL) / 256, 256, 0, stream>>>(w_qkv, wqh, QKV_N * D_MODEL);
    f32_to_f16_kernel<<<(D_MODEL * D_MODEL) / 256, 256, 0, stream>>>(w_out, woh, D_MODEL * D_MODEL);
    rmsnorm_kernel<<<NROWS, 256, 0, stream>>>(x, gamma, xn);
    gemm_wt_kernel<<<(NROWS / 128) * (QKV_N / 128), 256, 0, stream>>>(
        xn, wqh, nullptr, qkvh, NROWS, QKV_N, D_MODEL, QKV_N / 128);
    split_rope_kernel<<<(NROWS * N_HEAD * HEAD_DIM) / 256, 256, 0, stream>>>(qkvh, Qb, Kb, Vtb);
    attn_kernel<<<(BATCH * N_HEAD * (HW / 16)) / 8, 256, 0, stream>>>(Qb, Kb, Vtb, Ob);
    gemm_wt_kernel<<<(NROWS / 128) * (D_MODEL / 128), 256, 0, stream>>>(
        Ob, woh, (float*)d_out, nullptr, NROWS, D_MODEL, D_MODEL, D_MODEL / 128);
}
